// BiMambaWrapper_63479616635326
// MI455X (gfx1250) — compile-verified, matrix operand order checked by reading, not run on hardware
//
#include <hip/hip_runtime.h>
#include <hip/hip_bf16.h>

// ---------------- problem constants ----------------
#define DMODEL   768
#define DINNER   1536
#define DSTATE   16
#define DTRANK   48
#define XDBL_N   80            // DTRANK + 2*DSTATE
#define BB       2
#define LL       1024
#define NPOS     (BB*LL)       // 2048
#define NCH      8             // scan chunks
#define CHLEN    (LL/NCH)      // 128
#define DGROUPS  (DINNER/256)  // 6

typedef __attribute__((ext_vector_type(16))) _Float16 v16h;
typedef __attribute__((ext_vector_type(8)))  float    v8f;

// ---------------- helpers ----------------
__device__ __forceinline__ float siluf(float x) {
    return x / (1.0f + __expf(-x));
}
__device__ __forceinline__ float softplusf(float x) {
    return (x > 20.0f) ? x : log1pf(__expf(x));
}

// fragment loader: 16 halves = two aligned b128 loads (K-groups at p0, p1)
__device__ __forceinline__ v16h ldfrag(const _Float16* __restrict__ p0,
                                       const _Float16* __restrict__ p1) {
    union { uint4 q[2]; v16h v; } u;
    u.q[0] = *reinterpret_cast<const uint4*>(p0);
    u.q[1] = *reinterpret_cast<const uint4*>(p1);
    return u.v;
}
// fragment with second K-group zero (compile-time K tail)
__device__ __forceinline__ v16h ldfrag_lo(const _Float16* __restrict__ p0) {
    union { uint4 q[2]; v16h v; } u;
    u.q[0] = *reinterpret_cast<const uint4*>(p0);
    u.q[1] = make_uint4(0u, 0u, 0u, 0u);
    return u.v;
}

// ---------------- f32 -> f16 conversion (weights / activations) ------------
__global__ void __launch_bounds__(256)
cvt_f16(const float* __restrict__ s, _Float16* __restrict__ d, int n)
{
    int i = blockIdx.x * blockDim.x + threadIdx.x;
    if (i < n) d[i] = (_Float16)s[i];
}

// ---------------- WMMA GEMM:  C(MxN) = A(MxK) * B(NxK)^T  (f16 in, f32 acc)
// K compile-time (branch-free b128 loads; K%32==16 handled by constant tail).
// Each wave computes NREP adjacent 16x16 tiles along N, reusing the A
// fragment (NREP wmma per k-step).
// EPI==0: plain f32 store. EPI==1: softplus(acc + bias[n]). EPI==2: f32 + f16 copy.
template<int K, int NREP, int EPI>
__global__ void __launch_bounds__(256)
wmma_gemm(const _Float16* __restrict__ A, int lda,
          const _Float16* __restrict__ B, int ldb,
          float* __restrict__ C, int ldc,
          int M, int N,
          const float* __restrict__ bias,
          _Float16* __restrict__ Ch)
{
    static_assert(K % 32 == 0 || K % 32 == 16, "K tail must be 0 or 16");
    constexpr int KFULL = (K / 32) * 32;

    const int wave    = threadIdx.x >> 5;
    const int lane    = threadIdx.x & 31;
    const int ngroups = (N >> 4) / NREP;
    const int mtiles  = M >> 4;
    const int wid     = blockIdx.x * 8 + wave;
    if (wid >= mtiles * ngroups) return;
    const int tm  = wid / ngroups;
    const int tg  = wid % ngroups;
    const int row = lane & 15;   // A-row / B-col owned by this lane
    const int hi  = lane >> 4;   // K-group selector per ISA 16-bit layout

    const _Float16* arow = A + (size_t)(tm * 16 + row) * lda;
    const _Float16* brow = B + (size_t)(tg * NREP * 16 + row) * ldb;

    v8f acc[NREP];
#pragma unroll
    for (int r = 0; r < NREP; ++r) acc[r] = (v8f){};

    for (int kb = 0; kb < KFULL; kb += 32) {
        __builtin_prefetch(arow + kb + 256, 0, 1);
        const v16h av = ldfrag(arow + kb + hi * 8, arow + kb + 16 + hi * 8);
#pragma unroll
        for (int r = 0; r < NREP; ++r) {
            const _Float16* br = brow + (size_t)r * 16 * ldb;
            __builtin_prefetch(br + kb + 256, 0, 1);
            const v16h bv = ldfrag(br + kb + hi * 8, br + kb + 16 + hi * 8);
            acc[r] = __builtin_amdgcn_wmma_f32_16x16x32_f16(
                         false, av, false, bv, (short)0, acc[r], false, false);
        }
    }
    if constexpr (K % 32 != 0) {   // tail of 16: second K-group is zero
        const v16h av = ldfrag_lo(arow + KFULL + hi * 8);
#pragma unroll
        for (int r = 0; r < NREP; ++r) {
            const v16h bv = ldfrag_lo(brow + (size_t)r * 16 * ldb + KFULL + hi * 8);
            acc[r] = __builtin_amdgcn_wmma_f32_16x16x32_f16(
                         false, av, false, bv, (short)0, acc[r], false, false);
        }
    }

    // D layout: lane owns col = tile*16 + (lane&15); VGPR v holds row hi*8+v
#pragma unroll
    for (int r = 0; r < NREP; ++r) {
        const int col = (tg * NREP + r) * 16 + row;
#pragma unroll
        for (int v = 0; v < 8; ++v) {
            const int rr = tm * 16 + hi * 8 + v;
            float val = acc[r][v];
            if (EPI == 1) val = softplusf(val + bias[col]);
            C[(size_t)rr * ldc + col] = val;
            if (EPI == 2) Ch[(size_t)rr * ldc + col] = (_Float16)val;
        }
    }
}

// ---------------- depthwise causal conv (K=4) + SiLU, both directions -------
// u layout: (dir, B, L, DINNER); dir=1 stored in reversed-time coordinates.
// Writes f32 master (scan) and f16 mirror (x_proj GEMM A-operand).
__global__ void __launch_bounds__(256)
conv_silu(const float* __restrict__ xz,
          const float* __restrict__ wf, const float* __restrict__ bf,
          const float* __restrict__ wr, const float* __restrict__ br,
          float* __restrict__ u, _Float16* __restrict__ uh)
{
    int i = blockIdx.x * blockDim.x + threadIdx.x;
    if (i >= BB * LL * DINNER) return;
    const int d = i % DINNER;
    const int t = (i / DINNER) % LL;
    const int b = i / (DINNER * LL);

    // forward direction
    float acc = bf[d];
#pragma unroll
    for (int k = 0; k < 4; ++k) {
        const int tt = t - 3 + k;
        if (tt >= 0) acc += wf[d * 4 + k] * xz[((size_t)b * LL + tt) * (2 * DINNER) + d];
    }
    const size_t fidx = ((size_t)b * LL + t) * DINNER + d;
    const float uf = siluf(acc);
    u[fidx]  = uf;
    uh[fidx] = (_Float16)uf;

    // reverse direction (rev-time t maps to original time LL-1-t)
    float accr = br[d];
#pragma unroll
    for (int k = 0; k < 4; ++k) {
        const int tt = t - 3 + k;
        if (tt >= 0) accr += wr[d * 4 + k] * xz[((size_t)b * LL + (LL - 1 - tt)) * (2 * DINNER) + d];
    }
    const size_t ridx = (((size_t)BB + b) * LL + t) * DINNER + d;
    const float ur = siluf(accr);
    u[ridx]  = ur;
    uh[ridx] = (_Float16)ur;
}

// ---------------- scan pass 1: chunk-local scans ----------------
// In-place on uy (reads u, writes y_local + u*D). Records chunk-final state
// h and cumulative product P of dA per (d,s).
__global__ void __launch_bounds__(256)
scan_local(const float* __restrict__ delta,
           const float* __restrict__ xdbl,
           float* __restrict__ uy,
           const float* __restrict__ Alog_f, const float* __restrict__ Alog_r,
           const float* __restrict__ Dp_f,   const float* __restrict__ Dp_r,
           float* __restrict__ hfin, float* __restrict__ pfin)
{
    __shared__ float sBC[2 * DSTATE];
    const int bid = blockIdx.x;
    const int dg  = bid % DGROUPS;
    const int c   = (bid / DGROUPS) % NCH;
    const int b   = (bid / (DGROUPS * NCH)) % BB;
    const int dir =  bid / (DGROUPS * NCH * BB);
    const int d   = dg * 256 + threadIdx.x;

    const float* Alog = dir ? Alog_r : Alog_f;
    const float  Dd   = (dir ? Dp_r : Dp_f)[d];

    float A[DSTATE], h[DSTATE], P[DSTATE];
#pragma unroll
    for (int s = 0; s < DSTATE; ++s) {
        A[s] = -__expf(Alog[d * DSTATE + s]);
        h[s] = 0.0f; P[s] = 1.0f;
    }

    const size_t rowbase = (size_t)(dir * BB + b) * LL;
    for (int tt = 0; tt < CHLEN; ++tt) {
        const int t = c * CHLEN + tt;
        __syncthreads();
        if (threadIdx.x < 2 * DSTATE)
            sBC[threadIdx.x] = xdbl[(rowbase + t) * XDBL_N + DTRANK + threadIdx.x];
        __syncthreads();
        const size_t idx = (rowbase + t) * DINNER + d;
        const float dlt = delta[idx];
        const float uu  = uy[idx];
        float y = 0.0f;
#pragma unroll
        for (int s = 0; s < DSTATE; ++s) {
            const float dA = __expf(dlt * A[s]);
            h[s] = dA * h[s] + dlt * sBC[s] * uu;
            P[s] *= dA;
            y += h[s] * sBC[DSTATE + s];
        }
        uy[idx] = y + uu * Dd;   // z-gate applied in pass 3, after correction
    }
    const size_t hidx = (((size_t)(dir * BB + b) * NCH + c) * DINNER + d) * DSTATE;
#pragma unroll
    for (int s = 0; s < DSTATE; ++s) { hfin[hidx + s] = h[s]; pfin[hidx + s] = P[s]; }
}

// ---------------- scan pass 2: sequential chain over chunks ----------------
// hfin[c] is replaced in-place by the state *entering* chunk c.
__global__ void __launch_bounds__(256)
scan_chain(float* __restrict__ hfin, const float* __restrict__ pfin)
{
    const int i = blockIdx.x * blockDim.x + threadIdx.x;
    if (i >= 2 * BB * DINNER * DSTATE) return;
    const int ds = i % (DINNER * DSTATE);
    const int db = i / (DINNER * DSTATE);     // dir*BB + b
    float cur = 0.0f;
    for (int c = 0; c < NCH; ++c) {
        const size_t idx = ((size_t)db * NCH + c) * (DINNER * DSTATE) + ds;
        const float hf = hfin[idx];
        const float p  = pfin[idx];
        hfin[idx] = cur;
        cur = p * cur + hf;
    }
}

// ---------------- scan pass 3: inter-chunk correction + z-gate -------------
// y_t += C_t . (prod_{tau<=t} dA_tau) * h_in ; then multiply by silu(z).
__global__ void __launch_bounds__(256)
scan_fix(const float* __restrict__ delta,
         const float* __restrict__ xdbl,
         const float* __restrict__ xz,
         float* __restrict__ uy,
         const float* __restrict__ Alog_f, const float* __restrict__ Alog_r,
         const float* __restrict__ hin)
{
    __shared__ float sC[DSTATE];
    const int bid = blockIdx.x;
    const int dg  = bid % DGROUPS;
    const int c   = (bid / DGROUPS) % NCH;
    const int b   = (bid / (DGROUPS * NCH)) % BB;
    const int dir =  bid / (DGROUPS * NCH * BB);
    const int d   = dg * 256 + threadIdx.x;

    const float* Alog = dir ? Alog_r : Alog_f;
    float A[DSTATE], h0[DSTATE], P[DSTATE];
    const size_t hidx = (((size_t)(dir * BB + b) * NCH + c) * DINNER + d) * DSTATE;
#pragma unroll
    for (int s = 0; s < DSTATE; ++s) {
        A[s]  = -__expf(Alog[d * DSTATE + s]);
        h0[s] = hin[hidx + s];
        P[s]  = 1.0f;
    }

    const size_t rowbase = (size_t)(dir * BB + b) * LL;
    for (int tt = 0; tt < CHLEN; ++tt) {
        const int t = c * CHLEN + tt;
        __syncthreads();
        if (threadIdx.x < DSTATE)
            sC[threadIdx.x] = xdbl[(rowbase + t) * XDBL_N + DTRANK + DSTATE + threadIdx.x];
        __syncthreads();
        const size_t idx = (rowbase + t) * DINNER + d;
        const float dlt = delta[idx];
        float corr = 0.0f;
#pragma unroll
        for (int s = 0; s < DSTATE; ++s) {
            P[s] *= __expf(dlt * A[s]);
            corr += sC[s] * P[s] * h0[s];
        }
        const int zt = dir ? (LL - 1 - t) : t;
        const float zz = xz[((size_t)b * LL + zt) * (2 * DINNER) + DINNER + d];
        uy[idx] = (uy[idx] + corr) * siluf(zz);
    }
}

// ---------------- combine:  ysum = y_f + flip_L(y_r)  (f16 for out GEMM) ---
__global__ void __launch_bounds__(256)
combine_y(const float* __restrict__ uy, _Float16* __restrict__ ysum)
{
    int i = blockIdx.x * blockDim.x + threadIdx.x;
    if (i >= BB * LL * DINNER) return;
    const int d = i % DINNER;
    const int t = (i / DINNER) % LL;
    const int b = i / (DINNER * LL);
    const float yf = uy[((size_t)b * LL + t) * DINNER + d];
    const float yr = uy[(((size_t)BB + b) * LL + (LL - 1 - t)) * DINNER + d];
    ysum[((size_t)b * LL + t) * DINNER + d] = (_Float16)(yf + yr);
}

// ---------------- driver ----------------
extern "C" void kernel_launch(void* const* d_in, const int* in_sizes, int n_in,
                              void* d_out, int out_size, void* d_ws, size_t ws_size,
                              hipStream_t stream)
{
    const float* x        = (const float*)d_in[0];   // (B,L,768)
    const float* in_w     = (const float*)d_in[1];   // (3072,768)
    const float* out_w    = (const float*)d_in[2];   // (768,1536)
    const float* conv_w_f = (const float*)d_in[3];
    const float* conv_b_f = (const float*)d_in[4];
    const float* xproj_f  = (const float*)d_in[5];   // (80,1536)
    const float* dt_w_f   = (const float*)d_in[6];   // (1536,48)
    const float* dt_b_f   = (const float*)d_in[7];
    const float* Alog_f   = (const float*)d_in[8];   // (1536,16)
    const float* Dp_f     = (const float*)d_in[9];
    const float* conv_w_r = (const float*)d_in[10];
    const float* conv_b_r = (const float*)d_in[11];
    const float* xproj_r  = (const float*)d_in[12];
    const float* dt_w_r   = (const float*)d_in[13];
    const float* dt_b_r   = (const float*)d_in[14];
    const float* Alog_r   = (const float*)d_in[15];
    const float* Dp_r     = (const float*)d_in[16];

    // ---- workspace layout: f32 region, then f16 region ----
    float* ws   = (float*)d_ws;
    float* xz   = ws;                                   // 2048*3072
    float* u    = xz   + (size_t)NPOS * (2 * DINNER);   // 2 * 2048*1536 (in-place -> y)
    float* xdbl = u    + (size_t)2 * NPOS * DINNER;     // 2 * 2048*80
    float* dl   = xdbl + (size_t)2 * NPOS * XDBL_N;     // 2 * 2048*1536
    float* hfin = dl   + (size_t)2 * NPOS * DINNER;     // 2*2*8*1536*16
    float* pfin = hfin + (size_t)2 * BB * NCH * DINNER * DSTATE;
    _Float16* hbase  = (_Float16*)(pfin + (size_t)2 * BB * NCH * DINNER * DSTATE);
    _Float16* xh     = hbase;                                   // 2048*768
    _Float16* in_wh  = xh     + (size_t)NPOS * DMODEL;          // 3072*768
    _Float16* out_wh = in_wh  + (size_t)2 * DINNER * DMODEL;    // 768*1536
    _Float16* xpjh   = out_wh + (size_t)DMODEL * DINNER;        // 2 * 80*1536
    _Float16* dtwh   = xpjh   + (size_t)2 * XDBL_N * DINNER;    // 2 * 1536*48
    _Float16* uh     = dtwh   + (size_t)2 * DINNER * DTRANK;    // 2 * 2048*1536
    _Float16* xdblh  = uh     + (size_t)2 * NPOS * DINNER;      // 2 * 2048*80
    _Float16* ysumh  = xdblh  + (size_t)2 * NPOS * XDBL_N;      // 2048*1536

    // 0) one-shot f32->f16 conversions (weights + input activations)
    {
        auto cvt = [&](const float* s, _Float16* d, int n) {
            cvt_f16<<<(n + 255) / 256, 256, 0, stream>>>(s, d, n);
        };
        cvt(x,       xh,     NPOS * DMODEL);
        cvt(in_w,    in_wh,  2 * DINNER * DMODEL);
        cvt(out_w,   out_wh, DMODEL * DINNER);
        cvt(xproj_f, xpjh,                    XDBL_N * DINNER);
        cvt(xproj_r, xpjh + XDBL_N * DINNER,  XDBL_N * DINNER);
        cvt(dt_w_f,  dtwh,                    DINNER * DTRANK);
        cvt(dt_w_r,  dtwh + DINNER * DTRANK,  DINNER * DTRANK);
    }

    // 1) xz = x @ in_wT      (2048 x 3072, K=768), 4 N-tiles per wave
    {
        const int waves = (NPOS / 16) * ((2 * DINNER) / 16 / 4);
        wmma_gemm<DMODEL, 4, 0><<<(waves + 7) / 8, 256, 0, stream>>>(
            xh, DMODEL, in_wh, DMODEL, xz, 2 * DINNER, NPOS, 2 * DINNER,
            nullptr, nullptr);
    }
    // 2) u = silu(causal_conv(u_half)), both directions (f32 + f16 mirror)
    conv_silu<<<(BB * LL * DINNER + 255) / 256, 256, 0, stream>>>(
        xz, conv_w_f, conv_b_f, conv_w_r, conv_b_r, u, uh);
    // 3) x_dbl = u @ x_projT (2048 x 80, K=1536), f32 + f16 mirror
    for (int dir = 0; dir < 2; ++dir) {
        const int waves = (NPOS / 16) * (XDBL_N / 16 / 5);
        wmma_gemm<DINNER, 5, 2><<<(waves + 7) / 8, 256, 0, stream>>>(
            uh + (size_t)dir * NPOS * DINNER, DINNER,
            xpjh + (size_t)dir * XDBL_N * DINNER, DINNER,
            xdbl + (size_t)dir * NPOS * XDBL_N, XDBL_N,
            NPOS, XDBL_N, nullptr,
            xdblh + (size_t)dir * NPOS * XDBL_N);
    }
    // 4) delta = softplus(dt @ dt_wT + dt_b)  (2048 x 1536, K=48)
    for (int dir = 0; dir < 2; ++dir) {
        const int waves = (NPOS / 16) * (DINNER / 16 / 4);
        wmma_gemm<DTRANK, 4, 1><<<(waves + 7) / 8, 256, 0, stream>>>(
            xdblh + (size_t)dir * NPOS * XDBL_N, XDBL_N,
            dtwh + (size_t)dir * DINNER * DTRANK, DTRANK,
            dl + (size_t)dir * NPOS * DINNER, DINNER,
            NPOS, DINNER,
            dir ? dt_b_r : dt_b_f, nullptr);
    }
    // 5) chunked selective scan (3 passes)
    scan_local<<<2 * BB * NCH * DGROUPS, 256, 0, stream>>>(
        dl, xdbl, u, Alog_f, Alog_r, Dp_f, Dp_r, hfin, pfin);
    scan_chain<<<(2 * BB * DINNER * DSTATE + 255) / 256, 256, 0, stream>>>(hfin, pfin);
    scan_fix<<<2 * BB * NCH * DGROUPS, 256, 0, stream>>>(
        dl, xdbl, xz, u, Alog_f, Alog_r, hfin);
    // 6) ysum = y_f + flip(y_r)   (out_proj is shared & linear -> single GEMM)
    combine_y<<<(BB * LL * DINNER + 255) / 256, 256, 0, stream>>>(u, ysumh);
    // 7) out = ysum @ out_wT  (2048 x 768, K=1536)
    {
        const int waves = (NPOS / 16) * (DMODEL / 16 / 4);
        wmma_gemm<DINNER, 4, 0><<<(waves + 7) / 8, 256, 0, stream>>>(
            ysumh, DINNER, out_wh, DINNER, (float*)d_out, DMODEL,
            NPOS, DMODEL, nullptr, nullptr);
    }
}
